// DecoderStack_22308060136061
// MI455X (gfx1250) — compile-verified, matrix operand order checked by reading, not run on hardware
//
#include <hip/hip_runtime.h>
#include <hip/hip_bf16.h>

typedef __attribute__((ext_vector_type(16))) _Float16 v16h;
typedef __attribute__((ext_vector_type(8)))  _Float16 v8h;
typedef __attribute__((ext_vector_type(8)))  float    v8f;

#if __has_builtin(__builtin_amdgcn_global_load_async_to_lds_b128) && \
    __has_builtin(__builtin_amdgcn_s_wait_asynccnt)
#define ASYNC_COPY 1
typedef int v4i __attribute__((ext_vector_type(4)));
typedef __attribute__((address_space(1))) v4i* g4ptr;  // global int4*
typedef __attribute__((address_space(3))) v4i* l4ptr;  // LDS int4*
#else
#define ASYNC_COPY 0
#endif

// ---------------------------------------------------------------------------
// f32 -> f16 elementwise convert
// ---------------------------------------------------------------------------
__global__ __launch_bounds__(256) void cvt_f32_f16(const float* __restrict__ in,
                                                   _Float16* __restrict__ out, long n) {
    long i = (long)blockIdx.x * 256 + threadIdx.x;
    if (i < n) out[i] = (_Float16)in[i];
}

// ---------------------------------------------------------------------------
// Batched 32x32-tile transpose with convert-to-f16: out[c][r] = in[r][c]
// ---------------------------------------------------------------------------
template <typename TIn>
__global__ __launch_bounds__(256) void transpose_f16(const TIn* __restrict__ in,
                                                     _Float16* __restrict__ out,
                                                     long ldin, long ldout,
                                                     long sInB, long sOutB) {
    __shared__ float tile[32][33];
    const TIn* ip = in + (long)blockIdx.z * sInB;
    _Float16* op = out + (long)blockIdx.z * sOutB;
    int c0 = blockIdx.x * 32, r0 = blockIdx.y * 32;
    int tx = threadIdx.x & 31, ty = threadIdx.x >> 5;  // 32 x 8
#pragma unroll
    for (int i = 0; i < 32; i += 8)
        tile[ty + i][tx] = (float)ip[(long)(r0 + ty + i) * ldin + (c0 + tx)];
    __syncthreads();
#pragma unroll
    for (int i = 0; i < 32; i += 8)
        op[(long)(c0 + ty + i) * ldout + (r0 + tx)] = (_Float16)tile[tx][ty + i];
}

// ---------------------------------------------------------------------------
// WMMA GEMM: C[M,N] = alpha * A[M,K] * B^T  (+ bias, + relu)
//   A: f16 row-major (M x K), B: f16 row-major (N x K)  -> wmma-native layouts
//   256 threads = 8 wave32, wave grid 4x2, each wave owns (BM/4)x(BN/2) tile.
//   Double-buffered LDS; global->LDS via CDNA5 async copy when available.
//   Batched over blockIdx.z with (b,h) = (z/Hn, z%Hn) stride pairs.
// ---------------------------------------------------------------------------
template <int BM, int BN, int BK, bool OUTF16, int EPI>  // EPI 0:none 1:bias 2:bias+relu
__global__ __launch_bounds__(256) void wmma_gemm(
    const _Float16* __restrict__ A, long lda, long sAb, long sAh,
    const _Float16* __restrict__ B, long ldb, long sBb, long sBh,
    void* __restrict__ Cv, long ldc, long sCb, long sCh,
    int K, int Hn, const float* __restrict__ bias, float alpha) {
    constexpr int WM = BM / 4, WN = BN / 2;
    constexpr int MT = WM / 16, NT = WN / 16;
    constexpr int LDS_S = BK + 8;  // 40 halves = 80B rows (16B aligned)
    static_assert((BM * BK) % 2048 == 0 && (BN * BK) % 2048 == 0, "tile/loader mismatch");
    __shared__ _Float16 As[2][BM][LDS_S];
    __shared__ _Float16 Bs[2][BN][LDS_S];

    const int tid = threadIdx.x, lane = tid & 31, wave = tid >> 5;
    const int wm = (wave >> 1) * WM, wn = (wave & 1) * WN;
    const long m0 = (long)blockIdx.y * BM, n0 = (long)blockIdx.x * BN;
    const int zb = blockIdx.z / Hn, zh = blockIdx.z % Hn;
    const _Float16* Ab = A + zb * sAb + zh * sAh + m0 * lda;
    const _Float16* Bb = B + zb * sBb + zh * sBh + n0 * ldb;

    v8f acc[MT][NT] = {};

    constexpr int C8 = BK / 8;
    constexpr int AV = (BM * BK) / 2048;
    constexpr int BV = (BN * BK) / 2048;

    // per-thread staging coordinates
    int ra[AV], ca[AV], rb[BV], cb[BV];
#pragma unroll
    for (int i = 0; i < AV; i++) { int idx = tid + i * 256; ra[i] = idx / C8; ca[i] = (idx % C8) * 8; }
#pragma unroll
    for (int i = 0; i < BV; i++) { int idx = tid + i * 256; rb[i] = idx / C8; cb[i] = (idx % C8) * 8; }

#if ASYNC_COPY
    auto issue_tile = [&](int kk, int nb) {
#pragma unroll
        for (int i = 0; i < AV; i++)
            __builtin_amdgcn_global_load_async_to_lds_b128(
                (g4ptr)(Ab + (long)ra[i] * lda + kk + ca[i]),
                (l4ptr)&As[nb][ra[i]][ca[i]], 0, 0);
#pragma unroll
        for (int i = 0; i < BV; i++)
            __builtin_amdgcn_global_load_async_to_lds_b128(
                (g4ptr)(Bb + (long)rb[i] * ldb + kk + cb[i]),
                (l4ptr)&Bs[nb][rb[i]][cb[i]], 0, 0);
    };
    // prologue: tile 0
    issue_tile(0, 0);
    __builtin_amdgcn_s_wait_asynccnt(0);
    __syncthreads();
#else
    v8h ta[AV], tb[BV];
    auto fetch_tile = [&](int kk) {
#pragma unroll
        for (int i = 0; i < AV; i++) ta[i] = *(const v8h*)(Ab + (long)ra[i] * lda + kk + ca[i]);
#pragma unroll
        for (int i = 0; i < BV; i++) tb[i] = *(const v8h*)(Bb + (long)rb[i] * ldb + kk + cb[i]);
    };
    auto commit_tile = [&](int nb) {
#pragma unroll
        for (int i = 0; i < AV; i++) *(v8h*)&As[nb][ra[i]][ca[i]] = ta[i];
#pragma unroll
        for (int i = 0; i < BV; i++) *(v8h*)&Bs[nb][rb[i]][cb[i]] = tb[i];
    };
    fetch_tile(0);
    commit_tile(0);
    __syncthreads();
#endif

    int buf = 0;
    for (int k0 = 0; k0 < K; k0 += BK) {
        const int nbuf = buf ^ 1;
        const bool more = (k0 + BK) < K;
#if ASYNC_COPY
        if (more) issue_tile(k0 + BK, nbuf);  // overlaps with WMMAs below
#else
        if (more) fetch_tile(k0 + BK);        // global loads in flight during WMMAs
#endif

        // ISA 16-bit fragment layout: lanes0-15 K=[0..7],[16..23]; lanes16-31 K=[8..15],[24..31]
        const int rrow = lane & 15;
        const int chunk = (lane >> 4) * 8;
        v16h af[MT], bf[NT];
#pragma unroll
        for (int mt = 0; mt < MT; mt++) {
            const _Float16* p = &As[buf][wm + mt * 16 + rrow][chunk];
#pragma unroll
            for (int j = 0; j < 8; j++) { af[mt][j] = p[j]; af[mt][j + 8] = p[j + 16]; }
        }
#pragma unroll
        for (int nt = 0; nt < NT; nt++) {
            const _Float16* p = &Bs[buf][wn + nt * 16 + rrow][chunk];
#pragma unroll
            for (int j = 0; j < 8; j++) { bf[nt][j] = p[j]; bf[nt][j + 8] = p[j + 16]; }
        }
#pragma unroll
        for (int mt = 0; mt < MT; mt++)
#pragma unroll
            for (int nt = 0; nt < NT; nt++)
                acc[mt][nt] = __builtin_amdgcn_wmma_f32_16x16x32_f16(
                    false, af[mt], false, bf[nt], (short)0, acc[mt][nt], false, false);

        if (more) {
#if ASYNC_COPY
            __builtin_amdgcn_s_wait_asynccnt(0);
#else
            commit_tile(nbuf);
#endif
            __syncthreads();
        }
        buf = nbuf;
    }

    // Epilogue. C/D layout: VGPR j -> M=j (lanes 0-15) / M=j+8 (lanes 16-31), N=lane%16.
    float* Cf = (float*)Cv;
    _Float16* Ch = (_Float16*)Cv;
    const long cbase = zb * sCb + zh * sCh;
    const int rlo = (lane >> 4) * 8;
    const int ncol = lane & 15;
#pragma unroll
    for (int mt = 0; mt < MT; mt++)
#pragma unroll
        for (int nt = 0; nt < NT; nt++)
#pragma unroll
            for (int j = 0; j < 8; j++) {
                long m = m0 + wm + mt * 16 + rlo + j;
                long n = n0 + wn + nt * 16 + ncol;
                float v = acc[mt][nt][j] * alpha;
                if constexpr (EPI >= 1) v += bias[n];
                if constexpr (EPI == 2) v = fmaxf(v, 0.f);
                if constexpr (OUTF16) Ch[cbase + m * ldc + n] = (_Float16)v;
                else                  Cf[cbase + m * ldc + n] = v;
            }
}

// ---------------------------------------------------------------------------
// Softmax over the QUERY axis (t), faithful to reference (axis=2).
// scores stored [z=(b,h)][t][s] f16; each thread owns one s column (coalesced
// across lanes per t step). Online max/sum pass then in-place rewrite.
// ---------------------------------------------------------------------------
__global__ __launch_bounds__(256) void softmax_qaxis(_Float16* __restrict__ sc, int T_) {
    long z = blockIdx.z;
    int s = blockIdx.x * 256 + threadIdx.x;
    _Float16* p = sc + z * (long)T_ * T_ + s;
    float m = -3.4e38f, sum = 0.f;
    for (int t = 0; t < T_; t++) {
        float v = (float)p[(long)t * T_];
        if (v > m) { sum *= __expf(m - v); m = v; }
        sum += __expf(v - m);
    }
    float inv = 1.f / sum;
    for (int t = 0; t < T_; t++) {
        float v = (float)p[(long)t * T_];
        p[(long)t * T_] = (_Float16)(__expf(v - m) * inv);
    }
}

// ---------------------------------------------------------------------------
// out[row] = (a+b) - mean - unbiased_std, rows of 1024, 256 threads/row.
// ---------------------------------------------------------------------------
__global__ __launch_bounds__(256) void add_subnorm(const float* __restrict__ a,
                                                   const float* __restrict__ b,
                                                   float* __restrict__ out, int n) {
    long row = blockIdx.x;
    const float* ar = a + row * (long)n;
    const float* br = b + row * (long)n;
    float* orow = out + row * (long)n;
    float v[4], s = 0.f, ss = 0.f;
#pragma unroll
    for (int i = 0; i < 4; i++) {
        int idx = threadIdx.x + i * 256;
        float t = ar[idx] + br[idx];
        v[i] = t; s += t; ss += t * t;
    }
#pragma unroll
    for (int off = 16; off; off >>= 1) { s += __shfl_xor(s, off); ss += __shfl_xor(ss, off); }
    __shared__ float sb[8], ssb[8];
    int wave = threadIdx.x >> 5, lane = threadIdx.x & 31;
    if (!lane) { sb[wave] = s; ssb[wave] = ss; }
    __syncthreads();
    s = 0.f; ss = 0.f;
#pragma unroll
    for (int w = 0; w < 8; w++) { s += sb[w]; ss += ssb[w]; }
    float mean = s / n;
    float var = (ss - n * mean * mean) / (n - 1);
    float sd = sqrtf(fmaxf(var, 0.f));
#pragma unroll
    for (int i = 0; i < 4; i++) {
        int idx = threadIdx.x + i * 256;
        orow[idx] = v[i] - mean - sd;
    }
}

// ---------------------------------------------------------------------------
extern "C" void kernel_launch(void* const* d_in, const int* in_sizes, int n_in,
                              void* d_out, int out_size, void* d_ws, size_t ws_size,
                              hipStream_t stream) {
    (void)in_sizes; (void)n_in; (void)out_size; (void)ws_size;
    constexpr int Bc = 4, T = 1024, D = 1024, H = 16, DK = 64, FF = 4096;
    constexpr int BT = Bc * T;

    const float* x    = (const float*)d_in[0];
    const float* y    = (const float*)d_in[1];
    const float* Wq1  = (const float*)d_in[2];
    const float* Wk1  = (const float*)d_in[3];
    const float* Wv1  = (const float*)d_in[4];
    const float* Wo1  = (const float*)d_in[5];
    const float* Wq2  = (const float*)d_in[6];
    const float* Wk2  = (const float*)d_in[7];
    const float* Wv2  = (const float*)d_in[8];
    const float* Wo2  = (const float*)d_in[9];
    const float* W_in = (const float*)d_in[10];
    const float* b_in = (const float*)d_in[11];
    const float* W_out= (const float*)d_in[12];
    const float* b_out= (const float*)d_in[13];
    float* out = (float*)d_out;

    char* wsp = (char*)d_ws;
    auto alloc = [&](size_t bytes) -> void* {
        void* p = wsp; wsp += (bytes + 255) & ~(size_t)255; return p;
    };
    _Float16* yh    = (_Float16*)alloc((size_t)BT * D * 2);
    _Float16* xh    = (_Float16*)alloc((size_t)BT * D * 2);
    _Float16* WqT1  = (_Float16*)alloc((size_t)D * D * 2);
    _Float16* WkT1  = (_Float16*)alloc((size_t)D * D * 2);
    _Float16* WvT1  = (_Float16*)alloc((size_t)D * D * 2);
    _Float16* WoT1  = (_Float16*)alloc((size_t)D * D * 2);
    _Float16* WqT2  = (_Float16*)alloc((size_t)D * D * 2);
    _Float16* WkT2  = (_Float16*)alloc((size_t)D * D * 2);
    _Float16* WvT2  = (_Float16*)alloc((size_t)D * D * 2);
    _Float16* WoT2  = (_Float16*)alloc((size_t)D * D * 2);
    _Float16* Win16 = (_Float16*)alloc((size_t)FF * D * 2);
    _Float16* Wout16= (_Float16*)alloc((size_t)D * FF * 2);
    _Float16* Qh    = (_Float16*)alloc((size_t)BT * D * 2);
    _Float16* Kh    = (_Float16*)alloc((size_t)BT * D * 2);
    _Float16* Vh    = (_Float16*)alloc((size_t)BT * D * 2);
    _Float16* Vt    = (_Float16*)alloc((size_t)BT * D * 2);
    _Float16* scores= (_Float16*)alloc((size_t)Bc * H * T * T * 2);
    _Float16* part  = (_Float16*)alloc((size_t)BT * D * 2);
    float*    Obuf  = (float*)   alloc((size_t)BT * D * 4);
    float*    out1  = (float*)   alloc((size_t)BT * D * 4);
    float*    out2  = (float*)   alloc((size_t)BT * D * 4);
    _Float16* hidden= (_Float16*)alloc((size_t)BT * FF * 2);

    dim3 tb(256);
    // ---- converts ----
    cvt_f32_f16<<<(BT * D) / 256, tb, 0, stream>>>(y, yh, (long)BT * D);
    cvt_f32_f16<<<(BT * D) / 256, tb, 0, stream>>>(x, xh, (long)BT * D);
    cvt_f32_f16<<<(FF * D) / 256, tb, 0, stream>>>(W_in, Win16, (long)FF * D);
    cvt_f32_f16<<<(D * FF) / 256, tb, 0, stream>>>(W_out, Wout16, (long)D * FF);
    // ---- weight transposes to N x K f16 layouts ----
    // per-head: (D x DK) -> (DK x D), heads stacked along N
    transpose_f16<float><<<dim3(DK / 32, D / 32, H), tb, 0, stream>>>(Wq1, WqT1, DK, D, (long)D * DK, (long)DK * D);
    transpose_f16<float><<<dim3(DK / 32, D / 32, H), tb, 0, stream>>>(Wk1, WkT1, DK, D, (long)D * DK, (long)DK * D);
    transpose_f16<float><<<dim3(DK / 32, D / 32, H), tb, 0, stream>>>(Wv1, WvT1, DK, D, (long)D * DK, (long)DK * D);
    transpose_f16<float><<<dim3(DK / 32, D / 32, H), tb, 0, stream>>>(Wq2, WqT2, DK, D, (long)D * DK, (long)DK * D);
    transpose_f16<float><<<dim3(DK / 32, D / 32, H), tb, 0, stream>>>(Wk2, WkT2, DK, D, (long)D * DK, (long)DK * D);
    transpose_f16<float><<<dim3(DK / 32, D / 32, H), tb, 0, stream>>>(Wv2, WvT2, DK, D, (long)D * DK, (long)DK * D);
    transpose_f16<float><<<dim3(D / 32, D / 32, 1), tb, 0, stream>>>(Wo1, WoT1, D, D, 0, 0);
    transpose_f16<float><<<dim3(D / 32, D / 32, 1), tb, 0, stream>>>(Wo2, WoT2, D, D, 0, 0);

    auto run_mha = [&](const _Float16* qsrc, const _Float16* kvsrc,
                       _Float16* WqT, _Float16* WkT, _Float16* WvT, _Float16* WoT,
                       const float* resid, float* outf) {
        // projections: (BT x D) x (D x D)^N-major -> f16
        wmma_gemm<128, 128, 32, true, 0><<<dim3(D / 128, BT / 128, 1), tb, 0, stream>>>(
            qsrc, D, 0, 0, WqT, D, 0, 0, Qh, D, 0, 0, D, 1, nullptr, 1.f);
        wmma_gemm<128, 128, 32, true, 0><<<dim3(D / 128, BT / 128, 1), tb, 0, stream>>>(
            kvsrc, D, 0, 0, WkT, D, 0, 0, Kh, D, 0, 0, D, 1, nullptr, 1.f);
        wmma_gemm<128, 128, 32, true, 0><<<dim3(D / 128, BT / 128, 1), tb, 0, stream>>>(
            kvsrc, D, 0, 0, WvT, D, 0, 0, Vh, D, 0, 0, D, 1, nullptr, 1.f);
        // V^T per batch: (T x D) -> (D x T)
        transpose_f16<_Float16><<<dim3(D / 32, T / 32, Bc), tb, 0, stream>>>(
            Vh, Vt, D, T, (long)T * D, (long)D * T);
        // scores[z=(b,h)] = Q_bh (T x 64) * K_bh^T / 8 -> f16, batched over 64
        wmma_gemm<128, 128, 32, true, 0><<<dim3(T / 128, T / 128, Bc * H), tb, 0, stream>>>(
            Qh, D, (long)T * D, 64, Kh, D, (long)T * D, 64,
            scores, T, (long)H * T * T, (long)T * T, DK, H, nullptr, 0.125f);
        // softmax over query axis (faithful), in place
        softmax_qaxis<<<dim3(T / 256, 1, Bc * H), tb, 0, stream>>>(scores, T);
        // partial[b,t,h*64+v] = attn (T x T) * V_bh (T x 64)
        wmma_gemm<128, 64, 32, true, 0><<<dim3(1, T / 128, Bc * H), tb, 0, stream>>>(
            scores, T, (long)H * T * T, (long)T * T, Vt, T, (long)D * T, (long)64 * T,
            part, D, (long)T * D, 64, T, H, nullptr, 1.f);
        // output projection -> f32
        wmma_gemm<128, 128, 32, false, 0><<<dim3(D / 128, BT / 128, 1), tb, 0, stream>>>(
            part, D, 0, 0, WoT, D, 0, 0, Obuf, D, 0, 0, D, 1, nullptr, 1.f);
        add_subnorm<<<BT, tb, 0, stream>>>(Obuf, resid, outf, D);
    };

    // out1 = sub_norm(mha(y,y,y) + y)
    run_mha(yh, yh, WqT1, WkT1, WvT1, WoT1, y, out1);
    // out2 = sub_norm(mha(y,x,x) + out1)
    run_mha(yh, xh, WqT2, WkT2, WvT2, WoT2, out1, out2);
    // ff = relu(y @ W_in^T + b_in) @ W_out^T + b_out
    wmma_gemm<128, 128, 32, true, 2><<<dim3(FF / 128, BT / 128, 1), tb, 0, stream>>>(
        yh, D, 0, 0, Win16, D, 0, 0, hidden, FF, 0, 0, D, 1, b_in, 1.f);
    wmma_gemm<128, 128, 32, false, 1><<<dim3(D / 128, BT / 128, 1), tb, 0, stream>>>(
        hidden, FF, 0, 0, Wout16, FF, 0, 0, Obuf, D, 0, 0, FF, 1, b_out, 1.f);
    // out = sub_norm(ff + out2)
    add_subnorm<<<BT, tb, 0, stream>>>(Obuf, out2, out, D);
}
